// DeepSeekMoE_82068235092053
// MI455X (gfx1250) — matrix operand order repaired and checked by reading, not compiled, not run on hardware
//
#include <hip/hip_runtime.h>
#include <hip/hip_bf16.h>
#include <math.h>

// ---------------------------------------------------------------------------
// MoE: T=4096 tokens, H=1024, I=2048, E=16 experts, top-4.
// Sparse FLOPs ~137G, active weights 256MB f32 -> compute-bound => bf16 WMMA.
// GEMM2 A-tiles staged with CDNA5 async global->LDS copies (no conversion
// needed since hmid is already bf16).
// ---------------------------------------------------------------------------

#define E_ 16
#define K_ 4
#define B_ 4
#define S_ 1024
#define H_ 1024
#define I_ 2048
#define T_ (B_ * S_)   // 4096 tokens

typedef __attribute__((ext_vector_type(16))) __bf16        v16bf;
typedef __attribute__((ext_vector_type(8)))  float         v8f;
typedef __attribute__((ext_vector_type(4)))  unsigned int  v4u;
typedef __attribute__((ext_vector_type(2)))  unsigned int  v2u;

struct Frag32B { v4u q0, q1; };   // 32 bytes -> bit_cast to v16bf

// Pack two f32 -> packed bf16 pair (RNE). Prefer the HW packed convert.
static __device__ __forceinline__ unsigned pk2bf(float a, float b) {
#if __has_builtin(__builtin_amdgcn_cvt_pk_bf16_f32)
    auto pk = __builtin_amdgcn_cvt_pk_bf16_f32(a, b);
    return __builtin_bit_cast(unsigned, pk);
#else
    union { float f; unsigned u; } va, vb; va.f = a; vb.f = b;
    unsigned ra = va.u + 0x7FFFu + ((va.u >> 16) & 1u);
    unsigned rb = vb.u + 0x7FFFu + ((vb.u >> 16) & 1u);
    return (ra >> 16) | (rb & 0xFFFF0000u);
#endif
}

static __device__ __forceinline__ unsigned short f2bf(float f) {
    return (unsigned short)(pk2bf(f, 0.f) & 0xFFFFu);
}

// ---------------------------------------------------------------------------
// Router: one wave per token. Lanes 0..15 own expert e=lane (lane+16 computes
// the other half of the H-dot). Softmax + top-4 + renormalize via shuffles.
// ---------------------------------------------------------------------------
__global__ void moe_router(const float* __restrict__ x,
                           const float* __restrict__ rw,
                           const float* __restrict__ rb,
                           int*   __restrict__ counts,
                           int*   __restrict__ token_id,
                           int*   __restrict__ tkinfo,
                           float* __restrict__ tkw)
{
    const int wave = threadIdx.x >> 5;
    const int lane = threadIdx.x & 31;
    const int t = blockIdx.x * 8 + wave;
    if (t >= T_) return;

    const int e    = lane & 15;
    const int half = lane >> 4;
    const float* xt = x + (size_t)t * H_;

    float part = 0.f;
    const int h0 = half * (H_ / 2), h1 = h0 + (H_ / 2);
    for (int h = h0; h < h1; ++h)
        part = fmaf(xt[h], rw[h * E_ + e], part);
    part += __shfl_xor(part, 16, 32);          // full dot on all 32 lanes
    float logit = part + rb[e];

    // softmax over the 16 experts (width-16 butterfly keeps halves coherent)
    float mx = logit;
    for (int o = 8; o > 0; o >>= 1) mx = fmaxf(mx, __shfl_xor(mx, o, 16));
    float p = __expf(logit - mx);
    float s = p;
    for (int o = 8; o > 0; o >>= 1) s += __shfl_xor(s, o, 16);
    float prob = p / s;

    // top-4 with lowest-index tie-break (matches jax.lax.top_k)
    float rem = prob;
    float selw[K_]; int sele[K_];
    #pragma unroll
    for (int k = 0; k < K_; ++k) {
        float v = rem; int idx = e;
        for (int o = 8; o > 0; o >>= 1) {
            float ov = __shfl_xor(v, o, 16);
            int   oi = __shfl_xor(idx, o, 16);
            if (ov > v || (ov == v && oi < idx)) { v = ov; idx = oi; }
        }
        selw[k] = v; sele[k] = idx;
        if (e == idx) rem = -1.f;
    }
    const float inv = 1.f / (selw[0] + selw[1] + selw[2] + selw[3]);

    if (lane == 0) {
        #pragma unroll
        for (int k = 0; k < K_; ++k) {
            const int ee   = sele[k];
            const int slot = atomicAdd(&counts[ee], 1);
            token_id[ee * T_ + slot] = t;
            tkinfo[t * K_ + k] = (ee << 16) | slot;
            tkw[t * K_ + k]    = selw[k] * inv;
        }
    }
}

__global__ void moe_offsets(const int* __restrict__ counts,
                            int* __restrict__ offsets)
{
    if (threadIdx.x == 0) {
        int run = 0;
        for (int e = 0; e < E_; ++e) { offsets[e] = run; run += counts[e]; }
    }
}

// ---------------------------------------------------------------------------
// GEMM1 per expert: hmid = gelu(gather(X) @ w1[e] + b1[e]) -> bf16 rows.
// 128x128 tile per WG, 8 waves (4M x 2N), K-step 32, bf16 WMMA.
// ---------------------------------------------------------------------------
#define LDA 40   // pad 32->40 shorts: keeps 16B frag alignment, fewer bank conflicts

__global__ __launch_bounds__(256) void moe_gemm1(
    const float* __restrict__ x,  const float* __restrict__ w1,
    const float* __restrict__ b1,
    const int* __restrict__ counts, const int* __restrict__ offsets,
    const int* __restrict__ token_id,
    unsigned short* __restrict__ hmid)
{
    const int e   = blockIdx.z;
    const int cnt = counts[e];
    const int m0  = blockIdx.y * 128;
    if (m0 >= cnt) return;
    const int n0  = blockIdx.x * 128;

    __shared__ __align__(16) unsigned short As[128 * LDA];
    __shared__ __align__(16) unsigned short Bs[128 * LDA];   // Bs[n][k] (transposed)
    __shared__ int rows[128];

    const int tid = threadIdx.x;
    if (tid < 128) {
        const int m = m0 + tid;
        rows[tid] = (m < cnt) ? token_id[e * T_ + m] : -1;
    }
    __syncthreads();

    const int lane = tid & 31, w = tid >> 5;
    const int wm = w & 3, wn = w >> 2;        // 4 waves x 32 rows, 2 waves x 64 cols
    const int l15 = lane & 15, kh = lane >> 4;

    const v8f zf = {0.f,0.f,0.f,0.f,0.f,0.f,0.f,0.f};
    v8f acc[2][4];
    #pragma unroll
    for (int i = 0; i < 2; ++i)
        #pragma unroll
        for (int j = 0; j < 4; ++j) acc[i][j] = zf;

    const float* w1e = w1 + (size_t)e * H_ * I_;

    for (int k0 = 0; k0 < H_; k0 += 32) {
        if (k0) __syncthreads();              // prev compute done before restage
        // --- stage A: gathered X rows, f32 -> bf16, row-major [128][40]
        #pragma unroll
        for (int it = 0; it < 4; ++it) {
            const int idx = tid + it * 256;   // 128 rows x 8 quads
            const int r = idx >> 3, c = (idx & 7) << 2;
            const int trow = rows[r];
            float4 v = make_float4(0.f, 0.f, 0.f, 0.f);
            if (trow >= 0) v = *(const float4*)(x + (size_t)trow * H_ + k0 + c);
            *(v2u*)&As[r * LDA + c] = (v2u){pk2bf(v.x, v.y), pk2bf(v.z, v.w)};
        }
        // --- stage B: w1[e] f32 -> bf16, transposed to Bs[n][k]
        #pragma unroll
        for (int it = 0; it < 4; ++it) {
            const int idx = tid + it * 256;   // 32 k x 32 quad-cols
            const int kk = idx >> 5, nn = (idx & 31) << 2;
            const float4 v = *(const float4*)(w1e + (size_t)(k0 + kk) * I_ + n0 + nn);
            if (k0 + 32 < H_)
                __builtin_prefetch(w1e + (size_t)(k0 + 32 + kk) * I_ + n0 + nn, 0, 0);
            Bs[(nn + 0) * LDA + kk] = f2bf(v.x);
            Bs[(nn + 1) * LDA + kk] = f2bf(v.y);
            Bs[(nn + 2) * LDA + kk] = f2bf(v.z);
            Bs[(nn + 3) * LDA + kk] = f2bf(v.w);
        }
        __syncthreads();

        // --- fragments per ISA layout
        v16bf a[2], b[4];
        #pragma unroll
        for (int fm = 0; fm < 2; ++fm) {
            const unsigned short* p = &As[(wm * 32 + fm * 16 + l15) * LDA + kh * 8];
            Frag32B f; f.q0 = *(const v4u*)(p); f.q1 = *(const v4u*)(p + 16);
            a[fm] = __builtin_bit_cast(v16bf, f);
        }
        #pragma unroll
        for (int fn = 0; fn < 4; ++fn) {
            const unsigned short* p = &Bs[(wn * 64 + fn * 16 + l15) * LDA + kh * 8];
            Frag32B f; f.q0 = *(const v4u*)(p); f.q1 = *(const v4u*)(p + 16);
            b[fn] = __builtin_bit_cast(v16bf, f);
        }
        #pragma unroll
        for (int fm = 0; fm < 2; ++fm)
            #pragma unroll
            for (int fn = 0; fn < 4; ++fn)
                acc[fm][fn] = __builtin_amdgcn_wmma_f32_16x16x32_bf16(
                    false, a[fm], false, b[fn], (short)0, acc[fm][fn], false, false);
    }

    // --- epilogue: +b1, exact GELU, bf16 store to hmid rows
    const int off = offsets[e];
    #pragma unroll
    for (int fm = 0; fm < 2; ++fm)
        #pragma unroll
        for (int fn = 0; fn < 4; ++fn)
            #pragma unroll
            for (int j = 0; j < 8; ++j) {
                const int m = wm * 32 + fm * 16 + kh * 8 + j;
                if (m0 + m < cnt) {
                    const int col = n0 + wn * 64 + fn * 16 + l15;
                    const float v = acc[fm][fn][j] + b1[e * I_ + col];
                    const float g = 0.5f * v * (1.0f + erff(v * 0.70710678118654752440f));
                    hmid[(size_t)(off + m0 + m) * I_ + col] = f2bf(g);
                }
            }
}

// ---------------------------------------------------------------------------
// GEMM2 per expert: y = hmid @ w2[e]  (f32 rows; bias+gate applied in combine)
// A-tiles (bf16 hmid rows) are staged with async global->LDS b128 copies.
// ---------------------------------------------------------------------------
__global__ __launch_bounds__(256) void moe_gemm2(
    const unsigned short* __restrict__ hmid,
    const float* __restrict__ w2,
    const int* __restrict__ counts, const int* __restrict__ offsets,
    float* __restrict__ y)
{
    const int e   = blockIdx.z;
    const int cnt = counts[e];
    const int m0  = blockIdx.y * 128;
    if (m0 >= cnt) return;
    const int n0  = blockIdx.x * 128;
    const int off = offsets[e];

    __shared__ __align__(16) unsigned short As[128 * LDA];
    __shared__ __align__(16) unsigned short Bs[128 * LDA];

    const int tid = threadIdx.x, lane = tid & 31, w = tid >> 5;
    const int wm = w & 3, wn = w >> 2;
    const int l15 = lane & 15, kh = lane >> 4;

    const v8f zf = {0.f,0.f,0.f,0.f,0.f,0.f,0.f,0.f};
    v8f acc[2][4];
    #pragma unroll
    for (int i = 0; i < 2; ++i)
        #pragma unroll
        for (int j = 0; j < 4; ++j) acc[i][j] = zf;

    const float* w2e = w2 + (size_t)e * I_ * H_;

    // Per-thread A-copy geometry: 128 rows x 64B = 512 x 16B chunks, 2/thread.
    // Rows >= cnt are clamped to the last valid row (their outputs are never
    // stored), keeping every async address in-bounds.
    int arow[2], acol[2]; unsigned algs[2];
    #pragma unroll
    for (int it = 0; it < 2; ++it) {
        const int idx = tid + it * 256;       // 0..511
        const int r = idx >> 2, c8 = (idx & 3) << 3;   // c8 in shorts
        const int rv = (m0 + r < cnt) ? r : (cnt - 1 - m0);
        arow[it] = off + m0 + rv;
        acol[it] = c8;
        algs[it] = (unsigned)(size_t)&As[r * LDA + c8];   // LDS byte offset
    }

    for (int k0 = 0; k0 < I_; k0 += 32) {
        if (k0) __syncthreads();
        // --- stage A: async copy hmid bf16 rows straight into LDS
        #pragma unroll
        for (int it = 0; it < 2; ++it) {
            const unsigned long long g =
                (unsigned long long)(hmid + (size_t)arow[it] * I_ + k0 + acol[it]);
            asm volatile("global_load_async_to_lds_b128 %0, %1, off"
                         :: "v"(algs[it]), "v"(g) : "memory");
        }
        // --- stage B: w2[e] f32 -> bf16, transposed
        #pragma unroll
        for (int it = 0; it < 4; ++it) {
            const int idx = tid + it * 256;
            const int kk = idx >> 5, nn = (idx & 31) << 2;
            const float4 v = *(const float4*)(w2e + (size_t)(k0 + kk) * H_ + n0 + nn);
            if (k0 + 32 < I_)
                __builtin_prefetch(w2e + (size_t)(k0 + 32 + kk) * H_ + n0 + nn, 0, 0);
            Bs[(nn + 0) * LDA + kk] = f2bf(v.x);
            Bs[(nn + 1) * LDA + kk] = f2bf(v.y);
            Bs[(nn + 2) * LDA + kk] = f2bf(v.z);
            Bs[(nn + 3) * LDA + kk] = f2bf(v.w);
        }
        asm volatile("s_wait_asynccnt 0x0" ::: "memory");
        __syncthreads();

        v16bf a[2], b[4];
        #pragma unroll
        for (int fm = 0; fm < 2; ++fm) {
            const unsigned short* p = &As[(wm * 32 + fm * 16 + l15) * LDA + kh * 8];
            Frag32B f; f.q0 = *(const v4u*)(p); f.q1 = *(const v4u*)(p + 16);
            a[fm] = __builtin_bit_cast(v16bf, f);
        }
        #pragma unroll
        for (int fn = 0; fn < 4; ++fn) {
            const unsigned short* p = &Bs[(wn * 64 + fn * 16 + l15) * LDA + kh * 8];
            Frag32B f; f.q0 = *(const v4u*)(p); f.q1 = *(const v4u*)(p + 16);
            b[fn] = __builtin_bit_cast(v16bf, f);
        }
        #pragma unroll
        for (int fm = 0; fm < 2; ++fm)
            #pragma unroll
            for (int fn = 0; fn < 4; ++fn)
                acc[fm][fn] = __builtin_amdgcn_wmma_f32_16x16x32_bf16(
                    false, a[fm], false, b[fn], (short)0, acc[fm][fn], false, false);
    }

    #pragma unroll
    for (int fm = 0; fm < 2; ++fm)
        #pragma unroll
        for (int fn = 0; fn < 4; ++fn)
            #pragma unroll
            for (int j = 0; j < 8; ++j) {
                const int m = wm * 32 + fm * 16 + kh * 8 + j;
                if (m0 + m < cnt) {
                    const int col = n0 + wn * 64 + fn * 16 + l15;
                    y[(size_t)(off + m0 + m) * H_ + col] = acc[fm][fn][j];
                }
            }
}

// ---------------------------------------------------------------------------
// Combine: out[t] = sum_k wk * (y[row_k] + b2[e_k])   (fixed order => determ.)
// ---------------------------------------------------------------------------
__global__ void moe_combine(const float* __restrict__ y,
                            const float* __restrict__ b2,
                            const int* __restrict__ offsets,
                            const int* __restrict__ tkinfo,
                            const float* __restrict__ tkw,
                            float* __restrict__ out)
{
    const int t = blockIdx.x;
    int row[K_], ee[K_]; float wk[K_];
    #pragma unroll
    for (int k = 0; k < K_; ++k) {
        const int info = tkinfo[t * K_ + k];
        ee[k]  = info >> 16;
        row[k] = offsets[ee[k]] + (info & 0xFFFF);
        wk[k]  = tkw[t * K_ + k];
    }
    for (int h = threadIdx.x; h < H_; h += 256) {
        float a = 0.f;
        #pragma unroll
        for (int k = 0; k < K_; ++k)
            a += wk[k] * (y[(size_t)row[k] * H_ + h] + b2[ee[k] * H_ + h]);
        out[(size_t)t * H_ + h] = a;
    }
}

// ---------------------------------------------------------------------------
extern "C" void kernel_launch(void* const* d_in, const int* in_sizes, int n_in,
                              void* d_out, int out_size, void* d_ws, size_t ws_size,
                              hipStream_t stream)
{
    const float* x  = (const float*)d_in[0];
    const float* rw = (const float*)d_in[1];
    const float* rb = (const float*)d_in[2];
    const float* w1 = (const float*)d_in[3];
    const float* b1 = (const float*)d_in[4];
    const float* w2 = (const float*)d_in[5];
    const float* b2 = (const float*)d_in[6];
    float* out = (float*)d_out;

    char* ws = (char*)d_ws;
    int*   counts   = (int*)(ws);                                   // 16 ints
    int*   offsets  = (int*)(ws + 256);                             // 16 ints
    int*   token_id = (int*)(ws + 512);                             // E*T ints
    int*   tkinfo   = (int*)(ws + 512 + (size_t)E_ * T_ * 4);       // T*K ints
    float* tkw      = (float*)(ws + 512 + (size_t)E_ * T_ * 4 + (size_t)T_ * K_ * 4);
    size_t off_hmid = 512 + (size_t)E_ * T_ * 4 + 2 * (size_t)T_ * K_ * 4;
    off_hmid = (off_hmid + 255) & ~(size_t)255;
    unsigned short* hmid = (unsigned short*)(ws + off_hmid);        // (T*K) x I bf16
    size_t off_y = off_hmid + (size_t)T_ * K_ * I_ * 2;
    float* y = (float*)(ws + off_y);                                // (T*K) x H f32

    hipMemsetAsync(counts, 0, E_ * sizeof(int), stream);
    moe_router <<<T_ / 8, 256, 0, stream>>>(x, rw, rb, counts, token_id, tkinfo, tkw);
    moe_offsets<<<1, 32, 0, stream>>>(counts, offsets);
    moe_gemm1  <<<dim3(I_ / 128, T_ / 128, E_), 256, 0, stream>>>(
        x, w1, b1, counts, offsets, token_id, hmid);
    moe_gemm2  <<<dim3(H_ / 128, T_ / 128, E_), 256, 0, stream>>>(
        hmid, w2, counts, offsets, y);
    moe_combine<<<T_, 256, 0, stream>>>(y, b2, offsets, tkinfo, tkw, out);
}
